// DynamicSnakeConv_7980049236182
// MI455X (gfx1250) — matrix operand order repaired and checked by reading, not compiled, not run on hardware
//
#include <hip/hip_runtime.h>

// ---------------- problem constants ----------------
#define B_    4
#define CIN   128
#define COUT  128
#define H_    128
#define W_    128
#define HW    (H_ * W_)          // 16384
#define KDIM  (CIN * 9)          // 1152 = GEMM K
#define NTILES 8                 // COUT / 16
#define KSTEPS 36                // KDIM / 32
#define BPACK_ELEMS (NTILES * KSTEPS * 32 * 16)  // 147456 bf16
#define BPACK_BYTES (BPACK_ELEMS * 2)            // 294912 bytes

typedef __attribute__((ext_vector_type(8)))  __bf16 v8bf;
typedef __attribute__((ext_vector_type(16))) __bf16 v16bf;
typedef __attribute__((ext_vector_type(8)))  float  v8f;

// round-to-nearest-even float -> bf16 bits
__device__ __forceinline__ unsigned short f2bf(float f) {
    union { float f; unsigned int u; } v; v.f = f;
    unsigned int u = v.u;
    unsigned int r = (u + 0x7FFFu + ((u >> 16) & 1u)) >> 16;
    return (unsigned short)r;
}

// ---------------------------------------------------------------
// Kernel 1: repack w_main (O,C,3,3) f32 -> bf16 in WMMA-B fragment
// order.  B tile is 32x16 (KxN); lanes 0-15 hold K=ks..ks+15 for
// N=lane, lanes 16-31 hold K=ks+16..ks+31 for N=lane-16.  Each lane's
// 16 bf16 are stored contiguously so the GEMM loads them as 2x b128.
// Element index: ((nt*KSTEPS + step)*32 + lane)*16 + j
// ---------------------------------------------------------------
__global__ void pack_w_kernel(const float* __restrict__ w_main,
                              unsigned short* __restrict__ Bpack) {
    int tid = blockIdx.x * blockDim.x + threadIdx.x;
    if (tid >= NTILES * KSTEPS * 32) return;
    int lane = tid & 31;
    int step = (tid >> 5) % KSTEPS;
    int nt   = tid / (32 * KSTEPS);
    int row  = lane & 15;
    int hi   = lane >> 4;
    int N     = nt * 16 + row;                 // output channel
    int kbase = step * 32 + hi * 16;           // K = c*9 + tap
    unsigned short* dst = Bpack + (size_t)tid * 16;
#pragma unroll
    for (int j = 0; j < 16; ++j) {
        // w_main flat (O,C,3,3): o*1152 + (c*9 + tap) == o*KDIM + K
        dst[j] = f2bf(w_main[(size_t)N * KDIM + kbase + j]);
    }
}

// ---------------------------------------------------------------
// Kernel 2: offset conv (18 ch, 3x3, pad 1) + K,K cumsum + grid ->
// per-pixel, per-tap sample coordinates (ix,iy).
// Weight indices are lane-uniform -> scalar-cache loads.
// ---------------------------------------------------------------
__global__ void offset_grid_kernel(const float* __restrict__ x,
                                   const float* __restrict__ w_off,
                                   const float* __restrict__ b_off,
                                   float2* __restrict__ ixiy) {
    int p  = blockIdx.x * blockDim.x + threadIdx.x;   // 0..B*HW-1
    int b  = p >> 14;
    int hw = p & (HW - 1);
    int h  = hw >> 7;
    int w  = hw & (W_ - 1);

    float acc[18];
#pragma unroll
    for (int o = 0; o < 18; ++o) acc[o] = 0.0f;

    const float* xb = x + (size_t)b * CIN * HW;
    for (int c = 0; c < CIN; ++c) {
        const float* xc = xb + (size_t)c * HW;
        float xv[9];
#pragma unroll
        for (int t = 0; t < 9; ++t) {
            int dy = t / 3 - 1, dx = t % 3 - 1;
            int yy = h + dy, xx = w + dx;
            bool ok = (yy >= 0) && (yy < H_) && (xx >= 0) && (xx < W_);
            xv[t] = ok ? xc[yy * W_ + xx] : 0.0f;
        }
        const float* wb = w_off + c * 9;          // + o*CIN*9 + t
#pragma unroll
        for (int o = 0; o < 18; ++o) {
            const float* wo = wb + (size_t)o * CIN * 9;   // uniform addr
#pragma unroll
            for (int t = 0; t < 9; ++t)
                acc[o] = fmaf(xv[t], wo[t], acc[o]);
        }
    }

    float offv[18];
#pragma unroll
    for (int o = 0; o < 18; ++o) offv[o] = acc[o] + b_off[o];

    // base grid (x from w, y from h), linspace(-1,1,N)
    float gx = -1.0f + 2.0f * (float)w / (float)(W_ - 1);
    float gy = -1.0f + 2.0f * (float)h / (float)(H_ - 1);

    float2* dst = ixiy + (size_t)p * 9;
#pragma unroll
    for (int i = 0; i < 3; ++i) {
#pragma unroll
        for (int j = 0; j < 3; ++j) {
            float ox = 0.0f, oy = 0.0f;
#pragma unroll
            for (int ii = 0; ii <= i; ++ii) ox += offv[ii * 6 + j * 2 + 0];
#pragma unroll
            for (int jj = 0; jj <= j; ++jj) oy += offv[i * 6 + jj * 2 + 1];
            float sx = gx + ox, sy = gy + oy;
            float ix = ((sx + 1.0f) * (float)W_ - 1.0f) * 0.5f;
            float iy = ((sy + 1.0f) * (float)H_ - 1.0f) * 0.5f;
            dst[i * 3 + j] = make_float2(ix, iy);
        }
    }
}

// ---------------------------------------------------------------
// Kernel 3: fused bilinear-sample -> LDS (bf16) -> WMMA GEMM.
// Workgroup = 256 threads = 8 wave32; M-tile = 16 pixels; each wave
// owns one N-tile of 16 output channels; K = 1152 in 36 steps of 32.
// ---------------------------------------------------------------
__global__ __launch_bounds__(256)
void dsc_main_kernel(const float* __restrict__ x,
                     const unsigned short* __restrict__ Bpack,
                     const float2* __restrict__ ixiy,
                     const float* __restrict__ b_main,
                     float* __restrict__ out) {
    __shared__ unsigned short Ash[16 * KDIM];   // 36864 B, bf16 A panel

    int tid  = threadIdx.x;
    int wave = tid >> 5;
    int lane = tid & 31;
    int p0   = blockIdx.x * 16;                 // first pixel of M-tile
    int b    = p0 >> 14;                        // same batch for whole tile
    const float* xb = x + (size_t)b * CIN * HW;

    // ---- Phase 1: bilinear sampling into LDS -------------------
    // 144 (pos,tap) jobs; each lane handles 4 channels per job.
    for (int job = wave; job < 144; job += 8) {
        int pos = job / 9;
        int tap = job % 9;
        float2 g = ixiy[(size_t)(p0 + pos) * 9 + tap];
        float ix = g.x, iy = g.y;
        float fx0 = floorf(ix), fy0 = floorf(iy);
        int x0 = (int)fx0, y0 = (int)fy0;
        int x1 = x0 + 1,   y1 = y0 + 1;
        float wx1 = ix - fx0, wy1 = iy - fy0;
        float wx0 = 1.0f - wx1, wy0 = 1.0f - wy1;
        // fold validity into lerp weights; clamp indices so loads are in-bounds
        float vx0 = (x0 >= 0 && x0 < W_) ? 1.0f : 0.0f;
        float vx1 = (x1 >= 0 && x1 < W_) ? 1.0f : 0.0f;
        float vy0 = (y0 >= 0 && y0 < H_) ? 1.0f : 0.0f;
        float vy1 = (y1 >= 0 && y1 < H_) ? 1.0f : 0.0f;
        int x0c = min(max(x0, 0), W_ - 1), x1c = min(max(x1, 0), W_ - 1);
        int y0c = min(max(y0, 0), H_ - 1), y1c = min(max(y1, 0), H_ - 1);
        float m00 = wy0 * wx0 * vy0 * vx0;
        float m01 = wy0 * wx1 * vy0 * vx1;
        float m10 = wy1 * wx0 * vy1 * vx0;
        float m11 = wy1 * wx1 * vy1 * vx1;
        int i00 = y0c * W_ + x0c, i01 = y0c * W_ + x1c;
        int i10 = y1c * W_ + x0c, i11 = y1c * W_ + x1c;

        unsigned short* arow = Ash + pos * KDIM + tap;   // + c*9
#pragma unroll
        for (int cc = 0; cc < 4; ++cc) {
            int c = lane * 4 + cc;
            const float* xc = xb + (size_t)c * HW;       // L2-resident gathers
            float s = m00 * xc[i00] + m01 * xc[i01]
                    + m10 * xc[i10] + m11 * xc[i11];
            arow[c * 9] = f2bf(s);
        }
    }
    __syncthreads();

    // ---- Phase 2: WMMA GEMM (EXEC all-1s, fully uniform loop) --
    int nt  = wave;
    int row = lane & 15;
    int hi  = lane >> 4;

    float bias = b_main[nt * 16 + row];        // D element (M, N=lane&15)
    v8f acc;
#pragma unroll
    for (int j = 0; j < 8; ++j) acc[j] = bias;

    const unsigned short* asrc = Ash + row * KDIM;
    const unsigned short* bsrc = Bpack + ((size_t)(nt * KSTEPS) * 32 + lane) * 16;

    for (int s = 0; s < KSTEPS; ++s) {
        int ks = s * 32;
        // A: 16x32 layout matching packed B — lanes<16: K=ks..ks+15,
        //    lanes>=16: K=ks+16..ks+31  (two b128 LDS loads)
        v8bf a0 = *(const v8bf*)(asrc + ks + hi * 16);
        v8bf a1 = *(const v8bf*)(asrc + ks + hi * 16 + 8);
        // B: pre-packed fragment, two b128 global loads (L2 hit)
        v8bf b0 = *(const v8bf*)(bsrc);
        v8bf b1 = *(const v8bf*)(bsrc + 8);
        __builtin_prefetch(bsrc + 32 * 16 * 2, 0, 1);   // global_prefetch
        bsrc += 32 * 16;

        v16bf A, Bv;
#pragma unroll
        for (int i = 0; i < 8; ++i) {
            A[i] = a0[i];  A[8 + i] = a1[i];
            Bv[i] = b0[i]; Bv[8 + i] = b1[i];
        }
        acc = __builtin_amdgcn_wmma_f32_16x16x32_bf16(
                  /*neg_a=*/false, A, /*neg_b=*/false, Bv,
                  /*c_mod=*/(short)0, acc,
                  /*reuse_a=*/false, /*reuse_b=*/false);
    }

    // ---- write D: VGPR j -> M = hi*8 + j, N = lane&15 ----------
    int hwbase = p0 & (HW - 1);
    int o = nt * 16 + row;
    float* op = out + ((size_t)b * COUT + o) * HW + hwbase + hi * 8;
#pragma unroll
    for (int j = 0; j < 8; ++j) op[j] = acc[j];
}

// ---------------------------------------------------------------
extern "C" void kernel_launch(void* const* d_in, const int* in_sizes, int n_in,
                              void* d_out, int out_size, void* d_ws, size_t ws_size,
                              hipStream_t stream) {
    (void)in_sizes; (void)n_in; (void)out_size; (void)ws_size;
    const float* x      = (const float*)d_in[0];   // (4,128,128,128)
    const float* w_off  = (const float*)d_in[1];   // (18,128,3,3)
    const float* b_off  = (const float*)d_in[2];   // (18,)
    const float* w_main = (const float*)d_in[3];   // (128,128,3,3)
    const float* b_main = (const float*)d_in[4];   // (128,)
    float* out = (float*)d_out;                    // (4,128,128,128)

    unsigned short* Bpack = (unsigned short*)d_ws;                   // 288 KB
    float2* ixiy = (float2*)((char*)d_ws + BPACK_BYTES);             // 4.5 MB

    pack_w_kernel<<<(NTILES * KSTEPS * 32 + 255) / 256, 256, 0, stream>>>(w_main, Bpack);
    offset_grid_kernel<<<(B_ * HW) / 256, 256, 0, stream>>>(x, w_off, b_off, ixiy);
    dsc_main_kernel<<<(B_ * HW) / 16, 256, 0, stream>>>(x, Bpack, ixiy, b_main, out);
}
